// FFNwMoE_74380243632567
// MI455X (gfx1250) — compile-verified, matrix operand order checked by reading, not run on hardware
//
#include <hip/hip_runtime.h>
#include <stdint.h>

// ---------------- problem constants ----------------
#define T_TOK 8192
#define DDIM  2048
#define NEXP  8
#define TOPK  2
#define HDIM  1368
#define HPAD  1408   // HDIM padded up to multiple of 128 (and of 32 for WMMA K)
#define AUXC  0.01f

typedef __attribute__((ext_vector_type(16))) __bf16 v16bf;
typedef __attribute__((ext_vector_type(8)))  __bf16 v8bf;
typedef __attribute__((ext_vector_type(8)))  float  v8f;

__device__ __forceinline__ v8f wmma_bf16(v16bf a, v16bf b, v8f c) {
    return __builtin_amdgcn_wmma_f32_16x16x32_bf16(false, a, false, b, (short)0, c, false, false);
}

// Async DMA: 16B global -> LDS, tracked by ASYNCcnt (CDNA5 path, no VGPR staging).
__device__ __forceinline__ void async_b128(uint32_t lds_addr, const void* gaddr) {
    asm volatile("global_load_async_to_lds_b128 %0, %1, off"
                 :: "v"(lds_addr), "v"(gaddr) : "memory");
}
template<int N> __device__ __forceinline__ void wait_async() {
    asm volatile("s_wait_asynccnt %0" :: "i"(N));
}
__device__ __forceinline__ uint32_t lds_off(const void* p) {
    return (uint32_t)(uintptr_t)p;   // flat LDS mapping: addr[31:0] is the LDS byte address
}

// Load one 16x32 bf16 fragment from an LDS tile (row-major, strideH halfs,
// stride must be a multiple of 8 halfs for 16B alignment).
// CDNA5 16-bit A/B layout: lanes 0-15 -> rows 0-15 with K {0..7,16..23},
// lanes 16-31 -> rows 0-15 with K {8..15,24..31}.
__device__ __forceinline__ v16bf ldfrag(const __bf16* base, int strideH) {
    int l  = threadIdx.x & 31;
    int r  = l & 15;
    int hi = l >> 4;
    const __bf16* p = base + (size_t)r * strideH + hi * 8;
    union { v16bf v; v8bf h[2]; } u;
    u.h[0] = *(const v8bf*)(p);
    u.h[1] = *(const v8bf*)(p + 16);
    return u.v;
}

// ---------------- conversion kernels ----------------
__global__ __launch_bounds__(256) void conv_bf16(const float* __restrict__ src,
                                                 __bf16* __restrict__ dst, long n) {
    long i = (long)blockIdx.x * 256 + threadIdx.x;
    if (i < n) dst[i] = (__bf16)src[i];
}

// src rows of length HDIM -> dst rows of length HPAD (zero pad)
__global__ __launch_bounds__(256) void conv_pad_w2(const float* __restrict__ src,
                                                   __bf16* __restrict__ dst, long rows) {
    long i = (long)blockIdx.x * 256 + threadIdx.x;
    long total = rows * (long)HPAD;
    if (i >= total) return;
    long r = i / HPAD;
    int  c = (int)(i - r * HPAD);
    dst[i] = (c < HDIM) ? (__bf16)src[r * (long)HDIM + c] : (__bf16)0.0f;
}

// ---------------- router ----------------
__global__ __launch_bounds__(256)
void router_kernel(const float* __restrict__ x, const float* __restrict__ rw,
                   float* __restrict__ probs8, int* __restrict__ top1,
                   int* __restrict__ tki, float* __restrict__ tkw) {
    int wave = threadIdx.x >> 5, lane = threadIdx.x & 31;
    int t = blockIdx.x * 8 + wave;
    if (t >= T_TOK) return;
    float acc[NEXP];
#pragma unroll
    for (int e = 0; e < NEXP; ++e) acc[e] = 0.f;
    const float* xr = x + (size_t)t * DDIM;
    for (int d = lane; d < DDIM; d += 32) {
        float xv = xr[d];
#pragma unroll
        for (int e = 0; e < NEXP; ++e) acc[e] += xv * rw[e * DDIM + d];
    }
#pragma unroll
    for (int e = 0; e < NEXP; ++e)
        for (int off = 16; off; off >>= 1) acc[e] += __shfl_xor(acc[e], off, 32);
    if (lane == 0) {
        float mx = acc[0];
#pragma unroll
        for (int e = 1; e < NEXP; ++e) mx = acc[e] > mx ? acc[e] : mx;
        float p[NEXP], s = 0.f;
#pragma unroll
        for (int e = 0; e < NEXP; ++e) { p[e] = __expf(acc[e] - mx); s += p[e]; }
        float inv = 1.f / s;
#pragma unroll
        for (int e = 0; e < NEXP; ++e) p[e] *= inv;
        int i1 = 0;
#pragma unroll
        for (int e = 1; e < NEXP; ++e) if (p[e] > p[i1]) i1 = e;
        int i2 = (i1 == 0) ? 1 : 0;
#pragma unroll
        for (int e = 0; e < NEXP; ++e) if (e != i1 && p[e] > p[i2]) i2 = e;
#pragma unroll
        for (int e = 0; e < NEXP; ++e) probs8[(size_t)t * NEXP + e] = p[e];
        top1[t] = i1;
        tki[t * 2 + 0] = i1; tki[t * 2 + 1] = i2;
        tkw[t * 2 + 0] = p[i1]; tkw[t * 2 + 1] = p[i2];
    }
}

// ---------------- aux loss (deterministic reductions) ----------------
__global__ __launch_bounds__(256)
void aux_reduce(const float* __restrict__ probs8, const int* __restrict__ top1,
                float* __restrict__ probsum, float* __restrict__ density) {
    __shared__ float sS[256];
    __shared__ int   sC[256];
    int e = blockIdx.x;
    float s = 0.f; int c = 0;
    for (int t = threadIdx.x; t < T_TOK; t += 256) {
        s += probs8[(size_t)t * NEXP + e];
        c += (top1[t] == e);
    }
    sS[threadIdx.x] = s; sC[threadIdx.x] = c;
    __syncthreads();
    for (int o = 128; o; o >>= 1) {
        if (threadIdx.x < o) { sS[threadIdx.x] += sS[threadIdx.x + o]; sC[threadIdx.x] += sC[threadIdx.x + o]; }
        __syncthreads();
    }
    if (threadIdx.x == 0) { probsum[e] = sS[0]; density[e] = (float)sC[0]; }
}

__global__ void finalize_aux(const float* __restrict__ probsum,
                             const float* __restrict__ density,
                             float* __restrict__ outAux) {
    if (threadIdx.x == 0 && blockIdx.x == 0) {
        float a = 0.f;
        for (int e = 0; e < NEXP; ++e)
            a += (density[e] / (float)T_TOK) * (probsum[e] / (float)T_TOK);
        outAux[0] = AUXC * a * (float)NEXP;
    }
}

// ---------------- expert token lists ----------------
__global__ void init_counts(int* __restrict__ acnt) {
    if (threadIdx.x < NEXP && blockIdx.x == 0) acnt[threadIdx.x] = 0;
}

__global__ __launch_bounds__(256)
void build_lists(const int* __restrict__ tki, const float* __restrict__ tkw,
                 int* __restrict__ acnt, int* __restrict__ lTok,
                 int* __restrict__ lK, float* __restrict__ lW) {
    int t = blockIdx.x * 256 + threadIdx.x;
    if (t >= T_TOK) return;
#pragma unroll
    for (int k = 0; k < TOPK; ++k) {
        int e = tki[t * 2 + k];
        int s = atomicAdd(&acnt[e], 1);
        lTok[(size_t)e * T_TOK + s] = t;
        lK[(size_t)e * T_TOK + s]   = k;
        lW[(size_t)e * T_TOK + s]   = tkw[t * 2 + k];
    }
}

__global__ void scan_counts(const int* __restrict__ acnt, int* __restrict__ bases) {
    if (threadIdx.x == 0 && blockIdx.x == 0) {
        int o = 0;
        for (int e = 0; e < NEXP; ++e) { bases[e] = o; o += acnt[e]; }
    }
}

// ---------------- stage 1: hid = silu(x@w1^T) * (x@w3^T) ----------------
// block tile 64(M) x 128(N), K-step 32; 8 waves as 2(M)x4(N), wave tile 32x32
// Double-buffered LDS fed by global_load_async_to_lds_b128 (ASYNCcnt).
__global__ __launch_bounds__(256)
void ffn_stage1(const __bf16* __restrict__ xb,
                const __bf16* __restrict__ w1, const __bf16* __restrict__ w3,
                __bf16* __restrict__ hid,
                const int* __restrict__ listTok,
                const int* __restrict__ counts,
                const int* __restrict__ bases) {
    const int e   = blockIdx.z;
    const int cnt = counts ? counts[e] : T_TOK;
    const int m0  = blockIdx.x * 64;
    if (m0 >= cnt) return;
    const int n0   = blockIdx.y * 128;
    const int base = bases ? bases[e] : 0;

    __shared__ __bf16 sA [2][64  * 40];
    __shared__ __bf16 sB1[2][128 * 40];
    __shared__ __bf16 sB3[2][128 * 40];
    __shared__ int sTok[64];

    const int tid = threadIdx.x;
    if (tid < 64) {
        int s  = m0 + tid;
        int ss = (s < cnt) ? s : (cnt - 1);
        sTok[tid] = listTok ? listTok[(size_t)e * T_TOK + ss] : ss;
    }
    __syncthreads();

    const __bf16* w1e = w1 + (size_t)e * HDIM * DDIM;
    const __bf16* w3e = w3 + (size_t)e * HDIM * DDIM;

    v8f acc1[2][2] = {}; v8f acc3[2][2] = {};
    const int wave = tid >> 5;
    const int wm = wave & 1;   // M group (32 rows)
    const int wn = wave >> 1;  // N group (32 cols)
    const int arow = tid >> 2, aseg = tid & 3;
    const int br0 = tid >> 2,        bs0 = tid & 3;          // B chunk 0
    const int br1 = (tid + 256) >> 2, bs1 = tid & 3;         // B chunk 1
    int nr0 = n0 + br0; if (nr0 >= HDIM) nr0 = HDIM - 1;     // clamp pad region
    int nr1 = n0 + br1; if (nr1 >= HDIM) nr1 = HDIM - 1;

    // 5 async b128 per thread per buffer (in-order completion on ASYNCcnt)
    auto issue = [&](int buf, int k0) {
        async_b128(lds_off(&sA [buf][arow * 40 + aseg * 8]),
                   xb + (size_t)sTok[arow] * DDIM + k0 + aseg * 8);
        async_b128(lds_off(&sB1[buf][br0 * 40 + bs0 * 8]),
                   w1e + (size_t)nr0 * DDIM + k0 + bs0 * 8);
        async_b128(lds_off(&sB3[buf][br0 * 40 + bs0 * 8]),
                   w3e + (size_t)nr0 * DDIM + k0 + bs0 * 8);
        async_b128(lds_off(&sB1[buf][br1 * 40 + bs1 * 8]),
                   w1e + (size_t)nr1 * DDIM + k0 + bs1 * 8);
        async_b128(lds_off(&sB3[buf][br1 * 40 + bs1 * 8]),
                   w3e + (size_t)nr1 * DDIM + k0 + bs1 * 8);
    };

    const int nsteps = DDIM / 32;
    issue(0, 0);
    for (int s = 0; s < nsteps; ++s) {
        const int cur = s & 1;
        if (s + 1 < nsteps) { issue(cur ^ 1, (s + 1) * 32); wait_async<5>(); }
        else                { wait_async<0>(); }
        __syncthreads();
        v16bf a0 = ldfrag(&sA [cur][(wm * 32 +  0) * 40], 40);
        v16bf a1 = ldfrag(&sA [cur][(wm * 32 + 16) * 40], 40);
        v16bf p0 = ldfrag(&sB1[cur][(wn * 32 +  0) * 40], 40);
        v16bf p1 = ldfrag(&sB1[cur][(wn * 32 + 16) * 40], 40);
        v16bf q0 = ldfrag(&sB3[cur][(wn * 32 +  0) * 40], 40);
        v16bf q1 = ldfrag(&sB3[cur][(wn * 32 + 16) * 40], 40);
        acc1[0][0] = wmma_bf16(a0, p0, acc1[0][0]);
        acc1[0][1] = wmma_bf16(a0, p1, acc1[0][1]);
        acc1[1][0] = wmma_bf16(a1, p0, acc1[1][0]);
        acc1[1][1] = wmma_bf16(a1, p1, acc1[1][1]);
        acc3[0][0] = wmma_bf16(a0, q0, acc3[0][0]);
        acc3[0][1] = wmma_bf16(a0, q1, acc3[0][1]);
        acc3[1][0] = wmma_bf16(a1, q0, acc3[1][0]);
        acc3[1][1] = wmma_bf16(a1, q1, acc3[1][1]);
        __syncthreads();
    }

    const int lane = tid & 31;
    const int lr = (lane >> 4) * 8;
    const int lc = lane & 15;
#pragma unroll
    for (int i = 0; i < 2; ++i)
#pragma unroll
        for (int jx = 0; jx < 2; ++jx)
#pragma unroll
            for (int j = 0; j < 8; ++j) {
                int m = m0 + wm * 32 + i * 16 + lr + j;
                if (m < cnt) {
                    int n = n0 + wn * 32 + jx * 16 + lc;
                    float v1 = acc1[i][jx][j], v3 = acc3[i][jx][j];
                    float act = (n < HDIM) ? (v1 / (1.f + __expf(-v1))) * v3 : 0.f;
                    hid[(size_t)(base + m) * HPAD + n] = (__bf16)act;
                }
            }
}

// ---------------- stage 2: out_tile = (hid @ w2^T) * combine_weight ----------------
// block tile 128x128, K-step 32 (K = HPAD); 8 waves as 2(M)x4(N), wave tile 64x32
__global__ __launch_bounds__(256)
void ffn_stage2(const __bf16* __restrict__ hid,
                const __bf16* __restrict__ w2,    // [E][DDIM][HPAD]
                float* __restrict__ out,          // routed: [T][2][D]; shared: [T][D]
                const int* __restrict__ listTok,
                const int* __restrict__ listK,
                const float* __restrict__ listW,
                const int* __restrict__ counts,
                const int* __restrict__ bases,
                int routedMode) {
    const int e   = blockIdx.z;
    const int cnt = counts ? counts[e] : T_TOK;
    const int m0  = blockIdx.x * 128;
    if (m0 >= cnt) return;
    const int n0   = blockIdx.y * 128;
    const int base = bases ? bases[e] : 0;

    __shared__ __bf16 sA[2][128 * 40];
    __shared__ __bf16 sB[2][128 * 40];
    __shared__ int   sTok[128];
    __shared__ int   sKk[128];
    __shared__ float sW[128];
    __shared__ int   sHrow[128];

    const int tid = threadIdx.x;
    if (tid < 128) {
        int s  = m0 + tid;
        int ss = (s < cnt) ? s : (cnt - 1);
        sTok[tid]  = listTok ? listTok[(size_t)e * T_TOK + ss] : ss;
        sKk[tid]   = listK   ? listK  [(size_t)e * T_TOK + ss] : 0;
        sW[tid]    = listW   ? listW  [(size_t)e * T_TOK + ss] : 1.0f;
        sHrow[tid] = base + ss;
    }
    __syncthreads();

    const __bf16* w2e = w2 + (size_t)e * DDIM * HPAD;

    v8f acc[4][2] = {};
    const int wave = tid >> 5;
    const int wm = wave & 1;   // 64-row half
    const int wn = wave >> 1;  // 32-col group
    const int r0 = tid >> 2,          g0 = tid & 3;
    const int r1 = (tid + 256) >> 2,  g1 = tid & 3;

    // 4 async b128 per thread per buffer
    auto issue = [&](int buf, int k0) {
        async_b128(lds_off(&sA[buf][r0 * 40 + g0 * 8]),
                   hid + (size_t)sHrow[r0] * HPAD + k0 + g0 * 8);
        async_b128(lds_off(&sB[buf][r0 * 40 + g0 * 8]),
                   w2e + (size_t)(n0 + r0) * HPAD + k0 + g0 * 8);
        async_b128(lds_off(&sA[buf][r1 * 40 + g1 * 8]),
                   hid + (size_t)sHrow[r1] * HPAD + k0 + g1 * 8);
        async_b128(lds_off(&sB[buf][r1 * 40 + g1 * 8]),
                   w2e + (size_t)(n0 + r1) * HPAD + k0 + g1 * 8);
    };

    const int nsteps = HPAD / 32;
    issue(0, 0);
    for (int s = 0; s < nsteps; ++s) {
        const int cur = s & 1;
        if (s + 1 < nsteps) { issue(cur ^ 1, (s + 1) * 32); wait_async<4>(); }
        else                { wait_async<0>(); }
        __syncthreads();
        v16bf a0 = ldfrag(&sA[cur][(wm * 64 +  0) * 40], 40);
        v16bf a1 = ldfrag(&sA[cur][(wm * 64 + 16) * 40], 40);
        v16bf a2 = ldfrag(&sA[cur][(wm * 64 + 32) * 40], 40);
        v16bf a3 = ldfrag(&sA[cur][(wm * 64 + 48) * 40], 40);
        v16bf b0 = ldfrag(&sB[cur][(wn * 32 +  0) * 40], 40);
        v16bf b1 = ldfrag(&sB[cur][(wn * 32 + 16) * 40], 40);
        acc[0][0] = wmma_bf16(a0, b0, acc[0][0]);
        acc[0][1] = wmma_bf16(a0, b1, acc[0][1]);
        acc[1][0] = wmma_bf16(a1, b0, acc[1][0]);
        acc[1][1] = wmma_bf16(a1, b1, acc[1][1]);
        acc[2][0] = wmma_bf16(a2, b0, acc[2][0]);
        acc[2][1] = wmma_bf16(a2, b1, acc[2][1]);
        acc[3][0] = wmma_bf16(a3, b0, acc[3][0]);
        acc[3][1] = wmma_bf16(a3, b1, acc[3][1]);
        __syncthreads();
    }

    const int lane = tid & 31;
    const int lr = (lane >> 4) * 8;
    const int lc = lane & 15;
#pragma unroll
    for (int i = 0; i < 4; ++i)
#pragma unroll
        for (int jx = 0; jx < 2; ++jx)
#pragma unroll
            for (int j = 0; j < 8; ++j) {
                int rloc = wm * 64 + i * 16 + lr + j;
                int m = m0 + rloc;
                if (m < cnt) {
                    int n = n0 + wn * 32 + jx * 16 + lc;
                    float v = acc[i][jx][j] * sW[rloc];
                    size_t t = (size_t)sTok[rloc];
                    if (routedMode)
                        out[(t * 2 + sKk[rloc]) * (size_t)DDIM + n] = v;
                    else
                        out[t * (size_t)DDIM + n] = v;
                }
            }
}

// ---------------- combine: out = shared + routed slot0 + routed slot1 ----------------
__global__ __launch_bounds__(256)
void combine_kernel(float* __restrict__ out, const float* __restrict__ routed) {
    size_t i = (size_t)blockIdx.x * 256 + threadIdx.x;
    if (i >= (size_t)T_TOK * DDIM) return;
    size_t t = i / DDIM;
    size_t d = i - t * DDIM;
    out[i] += routed[(t * 2) * (size_t)DDIM + d] + routed[(t * 2 + 1) * (size_t)DDIM + d];
}

// ---------------- host launcher ----------------
extern "C" void kernel_launch(void* const* d_in, const int* in_sizes, int n_in,
                              void* d_out, int out_size, void* d_ws, size_t ws_size,
                              hipStream_t stream) {
    const float* x   = (const float*)d_in[0];
    const float* rw  = (const float*)d_in[1];
    const float* w1  = (const float*)d_in[2];
    const float* w2  = (const float*)d_in[3];
    const float* w3  = (const float*)d_in[4];
    const float* sw1 = (const float*)d_in[5];
    const float* sw2 = (const float*)d_in[6];
    const float* sw3 = (const float*)d_in[7];
    float* out = (float*)d_out;

    char* w = (char*)d_ws;
    size_t off = 0;
    auto take = [&](size_t bytes) -> void* {
        void* p = w + off;
        off = (off + bytes + 255) & ~(size_t)255;
        return p;
    };

    const size_t TD = (size_t)T_TOK * DDIM;
    __bf16* xb    = (__bf16*)take(TD * 2);
    __bf16* wb1   = (__bf16*)take((size_t)NEXP * HDIM * DDIM * 2);
    __bf16* wb3   = (__bf16*)take((size_t)NEXP * HDIM * DDIM * 2);
    __bf16* wb2   = (__bf16*)take((size_t)NEXP * DDIM * HPAD * 2);
    __bf16* swb1  = (__bf16*)take((size_t)HDIM * DDIM * 2);
    __bf16* swb3  = (__bf16*)take((size_t)HDIM * DDIM * 2);
    __bf16* swb2  = (__bf16*)take((size_t)DDIM * HPAD * 2);
    __bf16* hidR  = (__bf16*)take((size_t)T_TOK * TOPK * HPAD * 2);
    __bf16* hidS  = (__bf16*)take((size_t)T_TOK * HPAD * 2);
    float*  routedO = (float*)take((size_t)T_TOK * TOPK * DDIM * 4);
    float*  probs8  = (float*)take((size_t)T_TOK * NEXP * 4);
    int*    top1    = (int*)take((size_t)T_TOK * 4);
    int*    tki     = (int*)take((size_t)T_TOK * 2 * 4);
    float*  tkw     = (float*)take((size_t)T_TOK * 2 * 4);
    int*    lTok    = (int*)take((size_t)NEXP * T_TOK * 4);
    int*    lK      = (int*)take((size_t)NEXP * T_TOK * 4);
    float*  lW      = (float*)take((size_t)NEXP * T_TOK * 4);
    int*    acnt    = (int*)take(NEXP * 4);
    int*    bases   = (int*)take(NEXP * 4);
    float*  probsum = (float*)take(NEXP * 4);
    float*  density = (float*)take(NEXP * 4);
    (void)ws_size; (void)in_sizes; (void)n_in; (void)out_size;

    // 1) precision conversion (bf16 weights: ~150MB, resident in 192MB L2)
    {
        long n = (long)TD;
        conv_bf16<<<(unsigned)((n + 255) / 256), 256, 0, stream>>>(x, xb, n);
        n = (long)NEXP * HDIM * DDIM;
        conv_bf16<<<(unsigned)((n + 255) / 256), 256, 0, stream>>>(w1, wb1, n);
        conv_bf16<<<(unsigned)((n + 255) / 256), 256, 0, stream>>>(w3, wb3, n);
        n = (long)HDIM * DDIM;
        conv_bf16<<<(unsigned)((n + 255) / 256), 256, 0, stream>>>(sw1, swb1, n);
        conv_bf16<<<(unsigned)((n + 255) / 256), 256, 0, stream>>>(sw3, swb3, n);
        long rows = (long)NEXP * DDIM;
        long tot  = rows * HPAD;
        conv_pad_w2<<<(unsigned)((tot + 255) / 256), 256, 0, stream>>>(w2, wb2, rows);
        rows = (long)DDIM; tot = rows * HPAD;
        conv_pad_w2<<<(unsigned)((tot + 255) / 256), 256, 0, stream>>>(sw2, swb2, rows);
    }

    // 2) router + aux loss
    router_kernel<<<T_TOK / 8, 256, 0, stream>>>(x, rw, probs8, top1, tki, tkw);
    aux_reduce<<<NEXP, 256, 0, stream>>>(probs8, top1, probsum, density);
    finalize_aux<<<1, 64, 0, stream>>>(probsum, density, out + TD);

    // 3) expert gather lists
    init_counts<<<1, 64, 0, stream>>>(acnt);
    build_lists<<<T_TOK / 256, 256, 0, stream>>>(tki, tkw, acnt, lTok, lK, lW);
    scan_counts<<<1, 64, 0, stream>>>(acnt, bases);

    // 4) stage 1 (SwiGLU hidden), routed + shared
    {
        dim3 g1(T_TOK / 64, HPAD / 128, NEXP);
        ffn_stage1<<<g1, 256, 0, stream>>>(xb, wb1, wb3, hidR, lTok, acnt, bases);
        dim3 g1s(T_TOK / 64, HPAD / 128, 1);
        ffn_stage1<<<g1s, 256, 0, stream>>>(xb, swb1, swb3, hidS, nullptr, nullptr, nullptr);
    }

    // 5) stage 2 (down-proj): shared writes d_out directly, routed writes slots
    {
        dim3 g2s(T_TOK / 128, DDIM / 128, 1);
        ffn_stage2<<<g2s, 256, 0, stream>>>(hidS, swb2, out, nullptr, nullptr, nullptr,
                                            nullptr, nullptr, 0);
        dim3 g2(T_TOK / 128, DDIM / 128, NEXP);
        ffn_stage2<<<g2, 256, 0, stream>>>(hidR, wb2, routedO, lTok, lK, lW, acnt, bases, 1);
    }

    // 6) combine
    combine_kernel<<<(unsigned)((TD + 255) / 256), 256, 0, stream>>>(out, routedO);
}
